// LanguageModel_22952305230460
// MI455X (gfx1250) — matrix operand order repaired and checked by reading, not compiled, not run on hardware
//
#include <hip/hip_runtime.h>
#include <hip/hip_bf16.h>

typedef __bf16 bf16_t;
typedef __attribute__((ext_vector_type(16))) __bf16 v16bf;
typedef __attribute__((ext_vector_type(2)))  __bf16 bf16x2;
typedef __attribute__((ext_vector_type(8)))  float  v8f;

#define VOCAB   21128
#define HDIM    768
#define NLAYER  12
#define FDIM    3072
#define NHEAD   12
#define DHEAD   64
#define BATCH   32
#define SEQ     256
#define NROWS   (BATCH*SEQ)   /* 8192 */
#define CCHUNK  2048

static __device__ __forceinline__ unsigned short f2bf(float x) {
  return __builtin_bit_cast(unsigned short, (bf16_t)x);
}
static __device__ __forceinline__ void setpair(v16bf& f, int v, unsigned int w) {
  bf16x2 p = __builtin_bit_cast(bf16x2, w);
  f[2*v]   = p.x;
  f[2*v+1] = p.y;
}
static __device__ __forceinline__ unsigned ldsoff(const void* p) {
  // AS3 pointers cast to flat carry the LDS byte offset in the low 32 bits
  return (unsigned)(unsigned long long)(size_t)p;
}

// ---------------------------------------------------------------------------
// Generic bf16-WMMA GEMM, double-buffered LDS with register prefetch.
// out[M x Ncols] = A[M x K](fp32) @ Bw[K x ldb](fp32) + bias (+epilogue).
// M multiple of 64 (always 8192 here), K multiple of 32.
// EPI: 0 = bias, 1 = bias+exact GELU, 2 = bias+residual. OUTBF: bf16 output.
// ---------------------------------------------------------------------------
template<int EPI, bool OUTBF>
__global__ __launch_bounds__(256)
void gemm_kernel(const float* __restrict__ A, const float* __restrict__ Bw,
                 const float* __restrict__ bias, const float* __restrict__ resid,
                 void* __restrict__ outp, int M, int Ncols, int K, int ldb, int ldc)
{
  __shared__ __align__(16) unsigned short As[2][64][34];   // [m][k] bf16, padded
  __shared__ __align__(16) unsigned short Bs[2][128][34];  // [n][k] bf16, padded
  const int t    = threadIdx.x;
  const int m0   = blockIdx.y * 64;
  const int n0   = blockIdx.x * 128;
  const int wave = t >> 5, lane = t & 31;
  const int hf   = lane >> 4, nl = lane & 15;
  const int wm   = wave & 1,  wn = wave >> 1;              // 2 x 4 wave grid

  v8f c[2][2] = {};
  float ra[8], rb[16];
  const int nK = K >> 5;

  // prologue: tile 0 -> regs -> LDS buf 0
#pragma unroll
  for (int i = 0; i < 8; ++i) {
    int idx = i*256 + t; int m = idx >> 5, k = idx & 31;
    ra[i] = A[(size_t)(m0+m)*K + k];
  }
#pragma unroll
  for (int i = 0; i < 16; ++i) {
    int idx = i*256 + t; int k = idx >> 7, n = idx & 127;
    int col = n0 + n;
    rb[i] = (col < Ncols) ? Bw[(size_t)k*ldb + col] : 0.0f;
  }
#pragma unroll
  for (int i = 0; i < 8; ++i) {
    int idx = i*256 + t; int m = idx >> 5, k = idx & 31;
    As[0][m][k] = f2bf(ra[i]);
  }
#pragma unroll
  for (int i = 0; i < 16; ++i) {
    int idx = i*256 + t; int k = idx >> 7, n = idx & 127;
    Bs[0][n][k] = f2bf(rb[i]);
  }
  __syncthreads();

  for (int it = 0; it < nK; ++it) {
    const int p = it & 1;
    const int kk = (it + 1) << 5;
    if (it + 1 < nK) {                       // issue next tile's global loads
#pragma unroll
      for (int i = 0; i < 8; ++i) {
        int idx = i*256 + t; int m = idx >> 5, k = idx & 31;
        ra[i] = A[(size_t)(m0+m)*K + kk + k];
      }
#pragma unroll
      for (int i = 0; i < 16; ++i) {
        int idx = i*256 + t; int k = idx >> 7, n = idx & 127;
        int col = n0 + n;
        rb[i] = (col < Ncols) ? Bw[(size_t)(kk+k)*ldb + col] : 0.0f;
      }
    }
    // compute from buffer p (overlaps with the loads above)
    const unsigned int* Au = reinterpret_cast<const unsigned int*>(&As[p][0][0]); // pitch 17 u32
    const unsigned int* Bu = reinterpret_cast<const unsigned int*>(&Bs[p][0][0]);
    v16bf a0, a1, b0, b1;
#pragma unroll
    for (int v = 0; v < 8; ++v) {
      int ka = 2*v + 16*hf;                             // A-fragment K map (contiguous)
      int kb = 2*v + 16*hf;                             // B-fragment K map
      setpair(a0, v, Au[(wm*32 + nl)      * 17 + (ka >> 1)]);
      setpair(a1, v, Au[(wm*32 + 16 + nl) * 17 + (ka >> 1)]);
      setpair(b0, v, Bu[(wn*32 + nl)      * 17 + (kb >> 1)]);
      setpair(b1, v, Bu[(wn*32 + 16 + nl) * 17 + (kb >> 1)]);
    }
    c[0][0] = __builtin_amdgcn_wmma_f32_16x16x32_bf16(false, a0, false, b0, (short)0, c[0][0], false, false);
    c[0][1] = __builtin_amdgcn_wmma_f32_16x16x32_bf16(false, a0, false, b1, (short)0, c[0][1], false, false);
    c[1][0] = __builtin_amdgcn_wmma_f32_16x16x32_bf16(false, a1, false, b0, (short)0, c[1][0], false, false);
    c[1][1] = __builtin_amdgcn_wmma_f32_16x16x32_bf16(false, a1, false, b1, (short)0, c[1][1], false, false);
    if (it + 1 < nK) {                       // stage next tile into the other buffer
#pragma unroll
      for (int i = 0; i < 8; ++i) {
        int idx = i*256 + t; int m = idx >> 5, k = idx & 31;
        As[1-p][m][k] = f2bf(ra[i]);
      }
#pragma unroll
      for (int i = 0; i < 16; ++i) {
        int idx = i*256 + t; int k = idx >> 7, n = idx & 127;
        Bs[1-p][n][k] = f2bf(rb[i]);
      }
    }
    __syncthreads();
  }

#pragma unroll
  for (int mt = 0; mt < 2; ++mt)
#pragma unroll
    for (int nt = 0; nt < 2; ++nt)
#pragma unroll
      for (int r = 0; r < 8; ++r) {
        int row = m0 + wm*32 + mt*16 + r + 8*hf;
        int col = n0 + wn*32 + nt*16 + nl;
        if (col < Ncols) {
          float x = c[mt][nt][r] + bias[col];
          if (EPI == 1) x = 0.5f * x * (1.0f + erff(x * 0.70710678118f)); // exact GELU
          if (EPI == 2) x += resid[(size_t)row*ldc + col];
          if (OUTBF) ((unsigned short*)outp)[(size_t)row*ldc + col] = f2bf(x);
          else       ((float*)outp)[(size_t)row*ldc + col] = x;
        }
      }
}

// ---------------------------------------------------------------------------
// Fused flash attention, one head-slice per block. 4 waves x 16 q-rows.
// K tile staged via GLOBAL_LOAD_ASYNC_TO_LDS_B128 (ASYNCcnt); V transposed
// manually. Q/K/V bf16 [NROWS][HDIM]; ctx out fp32. Prefix-LM mask from s1.
// ---------------------------------------------------------------------------
__global__ __launch_bounds__(128)
void attn_kernel(const unsigned short* __restrict__ qg, const unsigned short* __restrict__ kg,
                 const unsigned short* __restrict__ vg, float* __restrict__ ctx,
                 const int* __restrict__ s1p)
{
  __shared__ __align__(16) unsigned short Kt[32][72];      // [key][d]; 144B rows (16B aligned)
  __shared__ __align__(16) unsigned short Vt[64][34];      // [d][key] (transposed)
  __shared__ __align__(16) unsigned short Pt[4][16][34];   // per-wave P transpose buffer
  const int bid  = blockIdx.x;
  const int qblk = bid & 3;               // SEQ/64
  const int bh   = bid >> 2;
  const int b    = bh / NHEAD, hd = bh % NHEAD;
  const int t    = threadIdx.x;
  const int wave = t >> 5, lane = t & 31, hf = lane >> 4, nl = lane & 15;
  const int s1   = s1p[0];
  const int qbase = qblk*64 + wave*16;

  // Q fragments for this wave's 16 rows, held in registers whole kernel.
  const unsigned short* qp = qg + (size_t)(b*SEQ + qbase + nl)*HDIM + hd*DHEAD;
  v16bf qf[2];
#pragma unroll
  for (int v = 0; v < 8; ++v) {
    int ka = 2*v + 16*hf;
    setpair(qf[0], v, *(const unsigned int*)(qp + ka));
    setpair(qf[1], v, *(const unsigned int*)(qp + 32 + ka));
  }
  float mrow[8], lrow[8];
  v8f cf[4] = {};
#pragma unroll
  for (int r = 0; r < 8; ++r) { mrow[r] = -3.0e38f; lrow[r] = 0.0f; }

  for (int it = 0; it < 8; ++it) {            // 32 keys per iteration
    __syncthreads();
    // K tile: async DMA global->LDS, 16B per lane, 2 issues cover 32x64 bf16
#pragma unroll
    for (int i = 0; i < 2; ++i) {
      int idx  = i*128 + t;
      int key  = idx >> 3, part = idx & 7;
      unsigned long long ga = (unsigned long long)(size_t)
          (kg + (size_t)(b*SEQ + it*32 + key)*HDIM + hd*DHEAD + part*8);
      unsigned la = ldsoff(&Kt[key][part*8]);
      asm volatile("global_load_async_to_lds_b128 %0, %1, off"
                   :: "v"(la), "v"(ga) : "memory");
    }
    // V tile: manual transpose into [d][key]
#pragma unroll
    for (int i = 0; i < 16; ++i) {
      int idx = i*128 + t;
      int key = idx >> 6, d = idx & 63;
      Vt[d][key] = vg[(size_t)(b*SEQ + it*32 + key)*HDIM + hd*DHEAD + d];
    }
    asm volatile("s_wait_asynccnt 0x0" ::: "memory");
    __syncthreads();

    v8f sc[2];
#pragma unroll
    for (int kt = 0; kt < 2; ++kt) {
      v8f acc = {};
#pragma unroll
      for (int dc = 0; dc < 2; ++dc) {
        v16bf bk;
#pragma unroll
        for (int v = 0; v < 8; ++v)
          setpair(bk, v, *(const unsigned int*)(&Kt[kt*16 + nl][dc*32 + 2*v + 16*hf]));
        acc = __builtin_amdgcn_wmma_f32_16x16x32_bf16(false, qf[dc], false, bk, (short)0, acc, false, false);
      }
      sc[kt] = acc;
    }
    // scale + prefix-LM mask (additive -1e9, matching reference)
#pragma unroll
    for (int kt = 0; kt < 2; ++kt)
#pragma unroll
      for (int r = 0; r < 8; ++r) {
        int key = it*32 + kt*16 + nl;
        int qr  = qbase + r + 8*hf;
        float sv = sc[kt][r] * 0.125f;              // 1/sqrt(64)
        if (!((key < s1) || (key <= qr))) sv += -1.0e9f;
        sc[kt][r] = sv;
      }
    // online softmax; row stats live in the 16-lane half-group owning that row
#pragma unroll
    for (int r = 0; r < 8; ++r) {
      float mx = fmaxf(sc[0][r], sc[1][r]);
      mx = fmaxf(mx, __shfl_xor(mx, 1, 32));
      mx = fmaxf(mx, __shfl_xor(mx, 2, 32));
      mx = fmaxf(mx, __shfl_xor(mx, 4, 32));
      mx = fmaxf(mx, __shfl_xor(mx, 8, 32));
      float mnew = fmaxf(mrow[r], mx);
      float p0 = __expf(sc[0][r] - mnew);
      float p1 = __expf(sc[1][r] - mnew);
      float ls = p0 + p1;
      ls += __shfl_xor(ls, 1, 32);
      ls += __shfl_xor(ls, 2, 32);
      ls += __shfl_xor(ls, 4, 32);
      ls += __shfl_xor(ls, 8, 32);
      float resc = __expf(mrow[r] - mnew);
      lrow[r] = lrow[r]*resc + ls;
      mrow[r] = mnew;
      sc[0][r] = p0; sc[1][r] = p1;
#pragma unroll
      for (int nt = 0; nt < 4; ++nt) cf[nt][r] *= resc;
    }
    // transpose P through per-wave LDS into an A fragment (in-wave DS ordered)
#pragma unroll
    for (int kt = 0; kt < 2; ++kt)
#pragma unroll
      for (int r = 0; r < 8; ++r)
        Pt[wave][r + 8*hf][kt*16 + nl] = f2bf(sc[kt][r]);
    v16bf pA;
#pragma unroll
    for (int v = 0; v < 8; ++v) {
      int ka = 2*v + 16*hf;
      setpair(pA, v, *(const unsigned int*)(&Pt[wave][nl][ka]));
    }
#pragma unroll
    for (int nt = 0; nt < 4; ++nt) {
      v16bf bV;
#pragma unroll
      for (int v = 0; v < 8; ++v)
        setpair(bV, v, *(const unsigned int*)(&Vt[nt*16 + nl][2*v + 16*hf]));
      cf[nt] = __builtin_amdgcn_wmma_f32_16x16x32_bf16(false, pA, false, bV, (short)0, cf[nt], false, false);
    }
  }
#pragma unroll
  for (int r = 0; r < 8; ++r) {
    float inv = 1.0f / lrow[r];
    int row = b*SEQ + qbase + r + 8*hf;
#pragma unroll
    for (int nt = 0; nt < 4; ++nt)
      ctx[(size_t)row*HDIM + hd*DHEAD + nt*16 + nl] = cf[nt][r] * inv;
  }
}

// ---------------------------------------------------------------------------
// Row LayerNorm (H=768, one block per row, 3 elems/thread)
// ---------------------------------------------------------------------------
__global__ __launch_bounds__(256)
void ln_kernel(const float* __restrict__ in, const float* __restrict__ g,
               const float* __restrict__ bta, float* __restrict__ out)
{
  __shared__ float r1[256], r2[256];
  const int row = blockIdx.x, t = threadIdx.x;
  const float* x = in + (size_t)row*HDIM;
  float vals[3], s = 0.f, sq = 0.f;
#pragma unroll
  for (int i = 0; i < 3; ++i) { float v = x[t + i*256]; vals[i] = v; s += v; sq += v*v; }
  r1[t] = s; r2[t] = sq; __syncthreads();
  for (int off = 128; off > 0; off >>= 1) {
    if (t < off) { r1[t] += r1[t+off]; r2[t] += r2[t+off]; }
    __syncthreads();
  }
  float mean = r1[0] * (1.0f/768.0f);
  float var  = r2[0] * (1.0f/768.0f) - mean*mean;
  float inv  = rsqrtf(var + 1e-12f);
  float* o = out + (size_t)row*HDIM;
#pragma unroll
  for (int i = 0; i < 3; ++i) { int j = t + i*256; o[j] = (vals[i]-mean)*inv*g[j] + bta[j]; }
}

__global__ __launch_bounds__(256)
void embed_kernel(const int* __restrict__ x, const float* __restrict__ we,
                  const float* __restrict__ pe, const float* __restrict__ te,
                  float* __restrict__ out)
{
  const int row = blockIdx.x, t = threadIdx.x;
  const int s = row % SEQ;
  const int tok = x[row];
#pragma unroll
  for (int i = 0; i < 3; ++i) {
    int j = t + i*256;
    out[(size_t)row*HDIM + j] = we[(size_t)tok*HDIM + j] + pe[(size_t)s*HDIM + j] + te[j];
  }
}

__global__ void init_stats(float* rowm, float* rowsum, float* rowt, float* acc) {
  int i = blockIdx.x*blockDim.x + threadIdx.x;
  if (i < NROWS) { rowm[i] = -3.0e38f; rowsum[i] = 0.f; rowt[i] = 0.f; }
  if (i == 0) { acc[0] = 0.f; acc[1] = 0.f; }
}

// online logsumexp over a vocab chunk; one block per token row
__global__ __launch_bounds__(256)
void lse_kernel(const float* __restrict__ lch, const int* __restrict__ y,
                float* __restrict__ rowm, float* __restrict__ rowsum,
                float* __restrict__ rowt, int c0, int nc)
{
  __shared__ float red[256];
  const int row = blockIdx.x, t = threadIdx.x;
  const float* lr = lch + (size_t)row*CCHUNK;
  float mx = -3.0e38f;
  for (int j = t; j < nc; j += 256) mx = fmaxf(mx, lr[j]);
  red[t] = mx; __syncthreads();
  for (int off = 128; off > 0; off >>= 1) {
    if (t < off) red[t] = fmaxf(red[t], red[t+off]);
    __syncthreads();
  }
  float mold = rowm[row];
  float mnew = fmaxf(mold, red[0]);
  __syncthreads();
  float s = 0.f;
  for (int j = t; j < nc; j += 256) s += __expf(lr[j] - mnew);
  red[t] = s; __syncthreads();
  for (int off = 128; off > 0; off >>= 1) {
    if (t < off) red[t] += red[t+off];
    __syncthreads();
  }
  if (t == 0) {
    rowsum[row] = rowsum[row]*__expf(mold - mnew) + red[0];
    rowm[row] = mnew;
    int loc = y[row] - c0;
    if (loc >= 0 && loc < nc) rowt[row] = lr[loc];
  }
}

__global__ void loss_kernel(const int* __restrict__ y, const float* __restrict__ rowm,
                            const float* __restrict__ rowsum, const float* __restrict__ rowt,
                            float* __restrict__ acc)
{
  int i = blockIdx.x*blockDim.x + threadIdx.x;
  if (i >= NROWS) return;
  bool valid = (y[i] != -100);
  float nll = rowm[i] + logf(rowsum[i]) - rowt[i];
  atomicAdd(&acc[0], valid ? nll : 0.0f);
  atomicAdd(&acc[1], valid ? 1.0f : 0.0f);
}

__global__ void finalize_kernel(const float* __restrict__ acc, float* __restrict__ out) {
  out[0] = acc[0] / acc[1];
}

// ---------------------------------------------------------------------------
extern "C" void kernel_launch(void* const* d_in, const int* in_sizes, int n_in,
                              void* d_out, int out_size, void* d_ws, size_t ws_size,
                              hipStream_t stream)
{
  (void)in_sizes; (void)n_in; (void)out_size; (void)ws_size;
  const int*   x    = (const int*)d_in[0];
  const int*   y    = (const int*)d_in[1];
  const int*   s1p  = (const int*)d_in[2];
  const float* we   = (const float*)d_in[4];
  const float* pe   = (const float*)d_in[5];
  const float* te   = (const float*)d_in[6];
  const float* elng = (const float*)d_in[7];
  const float* elnb = (const float*)d_in[8];
  const float* Wq   = (const float*)d_in[9];
  const float* bq   = (const float*)d_in[10];
  const float* Wk   = (const float*)d_in[11];
  const float* bk   = (const float*)d_in[12];
  const float* Wv   = (const float*)d_in[13];
  const float* bv   = (const float*)d_in[14];
  const float* Wo   = (const float*)d_in[15];
  const float* bo   = (const float*)d_in[16];
  const float* ln1g = (const float*)d_in[17];
  const float* ln1b = (const float*)d_in[18];
  const float* W1   = (const float*)d_in[19];
  const float* b1   = (const float*)d_in[20];
  const float* W2   = (const float*)d_in[21];
  const float* b2   = (const float*)d_in[22];
  const float* ln2g = (const float*)d_in[23];
  const float* ln2b = (const float*)d_in[24];
  const float* clsW = (const float*)d_in[25];
  const float* clsb = (const float*)d_in[26];

  char* ws = (char*)d_ws;
  size_t off = 0;
  auto alloc = [&](size_t bytes) -> void* {
    void* p = ws + off;
    off += (bytes + 255) & ~(size_t)255;
    return p;
  };
  float*          h    = (float*)alloc((size_t)NROWS*HDIM*4);
  float*          tmp  = (float*)alloc((size_t)NROWS*HDIM*4);
  float*          ctx  = (float*)alloc((size_t)NROWS*HDIM*4);
  unsigned short* qbuf = (unsigned short*)alloc((size_t)NROWS*HDIM*2);
  unsigned short* kbuf = (unsigned short*)alloc((size_t)NROWS*HDIM*2);
  unsigned short* vbuf = (unsigned short*)alloc((size_t)NROWS*HDIM*2);
  float*          ffm  = (float*)alloc((size_t)NROWS*FDIM*4);   // reused for logit chunks
  float*          rowm = (float*)alloc((size_t)NROWS*4);
  float*          rsum = (float*)alloc((size_t)NROWS*4);
  float*          rowt = (float*)alloc((size_t)NROWS*4);
  float*          acc  = (float*)alloc(256);
  float*          lch  = ffm;   // NROWS*CCHUNK*4 <= NROWS*FDIM*4

  const dim3 gH((HDIM + 127)/128, NROWS/64);
  const dim3 gF((FDIM + 127)/128, NROWS/64);

  embed_kernel<<<NROWS, 256, 0, stream>>>(x, we, pe, te, tmp);
  ln_kernel<<<NROWS, 256, 0, stream>>>(tmp, elng, elnb, h);

  for (int l = 0; l < NLAYER; ++l) {
    const size_t wHH = (size_t)l*HDIM*HDIM;
    const size_t wHF = (size_t)l*HDIM*FDIM;
    gemm_kernel<0,true ><<<gH, 256, 0, stream>>>(h, Wq + wHH, bq + l*HDIM, nullptr, qbuf, NROWS, HDIM, HDIM, HDIM, HDIM);
    gemm_kernel<0,true ><<<gH, 256, 0, stream>>>(h, Wk + wHH, bk + l*HDIM, nullptr, kbuf, NROWS, HDIM, HDIM, HDIM, HDIM);
    gemm_kernel<0,true ><<<gH, 256, 0, stream>>>(h, Wv + wHH, bv + l*HDIM, nullptr, vbuf, NROWS, HDIM, HDIM, HDIM, HDIM);
    attn_kernel<<<BATCH*NHEAD*(SEQ/64), 128, 0, stream>>>(qbuf, kbuf, vbuf, ctx, s1p);
    gemm_kernel<2,false><<<gH, 256, 0, stream>>>(ctx, Wo + wHH, bo + l*HDIM, h, tmp, NROWS, HDIM, HDIM, HDIM, HDIM);
    ln_kernel<<<NROWS, 256, 0, stream>>>(tmp, ln1g + l*HDIM, ln1b + l*HDIM, h);
    gemm_kernel<1,false><<<gF, 256, 0, stream>>>(h, W1 + wHF, b1 + l*FDIM, nullptr, ffm, NROWS, FDIM, HDIM, FDIM, FDIM);
    gemm_kernel<2,false><<<gH, 256, 0, stream>>>(ffm, W2 + wHF, b2 + l*HDIM, h, tmp, NROWS, HDIM, FDIM, HDIM, HDIM);
    ln_kernel<<<NROWS, 256, 0, stream>>>(tmp, ln2g + l*HDIM, ln2b + l*HDIM, h);
  }

  init_stats<<<(NROWS + 255)/256, 256, 0, stream>>>(rowm, rsum, rowt, acc);
  for (int c0 = 0; c0 < VOCAB; c0 += CCHUNK) {
    int nc = VOCAB - c0; if (nc > CCHUNK) nc = CCHUNK;
    dim3 gV((nc + 127)/128, NROWS/64);
    gemm_kernel<0,false><<<gV, 256, 0, stream>>>(h, clsW + c0, clsb + c0, nullptr, lch, NROWS, nc, HDIM, VOCAB, CCHUNK);
    lse_kernel<<<NROWS, 256, 0, stream>>>(lch, y, rowm, rsum, rowt, c0, nc);
  }
  loss_kernel<<<(NROWS + 255)/256, 256, 0, stream>>>(y, rowm, rsum, rowt, acc);
  finalize_kernel<<<1, 1, 0, stream>>>(acc, (float*)d_out);
}
